// GINModelBatches_73521250173227
// MI455X (gfx1250) — compile-verified, matrix operand order checked by reading, not run on hardware
//
#include <hip/hip_runtime.h>

// ---------------------------------------------------------------------------
// GIN-on-MI455X: edge scatter is L2-resident memory work; all GEMMs go through
// v_wmma_f32_16x16x32_bf16 (bf16 A/B, f32 accum) with fused bias+activation.
// ---------------------------------------------------------------------------

typedef __attribute__((ext_vector_type(16))) __bf16 v16bf;
typedef __attribute__((ext_vector_type(8)))  __bf16 v8bf;
typedef __attribute__((ext_vector_type(4)))  __bf16 v4bf;
typedef __attribute__((ext_vector_type(8)))  float  v8f;

#define NPNODES 50000   // multiple of 16 (50000 = 3125*16)
#define HID     128
#define NGRAPH  64
#define NCLS    10
#define SLOPE   0.2f

static inline int cdivi(int a, int b) { return (a + b - 1) / b; }

// ---------------------------- utility kernels ------------------------------

__global__ __launch_bounds__(256) void k_zero4(float* __restrict__ p, int n4) {
    int t = blockIdx.x * 256 + threadIdx.x;
    if (t < n4) ((float4*)p)[t] = make_float4(0.f, 0.f, 0.f, 0.f);
}

// Wt[n][k] = bf16(W[k][n]); rows n >= N zero-padded (for Wp2: 10 -> 16)
__global__ __launch_bounds__(256)
void k_pack_wt(const float* __restrict__ W, __bf16* __restrict__ Wt,
               int K, int N, int Npad) {
    int t = blockIdx.x * 256 + threadIdx.x;
    if (t >= Npad * K) return;
    int n = t / K, k = t % K;
    float v = (n < N) ? W[(size_t)k * N + n] : 0.0f;
    Wt[(size_t)n * K + k] = (__bf16)v;
}

__global__ void k_pad_bias(const float* __restrict__ b, float* __restrict__ out,
                           int n, int npad) {
    int t = threadIdx.x;
    if (t < npad) out[t] = (t < n) ? b[t] : 0.0f;
}

// ---------------------------- edge aggregation -----------------------------

// One wave covers one edge's feature row (F/4 float4 chunks, coalesced).
__global__ __launch_bounds__(256)
void k_scatter(const float* __restrict__ x, const int* __restrict__ src,
               const int* __restrict__ dst, float* __restrict__ agg,
               int E, int F) {
    int cpe = F >> 2;                       // float4 chunks per edge
    int t = blockIdx.x * 256 + threadIdx.x;
    if (t >= E * cpe) return;
    int e = t / cpe, c = t % cpe;
    int s = src[e], d = dst[e];
    float4 v = *(const float4*)(x + (size_t)s * F + c * 4);
    float* ap = agg + (size_t)d * F + c * 4;
    atomicAdd(ap + 0, v.x);
    atomicAdd(ap + 1, v.y);
    atomicAdd(ap + 2, v.z);
    atomicAdd(ap + 3, v.w);
}

// comb = bf16((2+eps)*x + agg)   (self-loop folded into the 2+eps term)
__global__ __launch_bounds__(256)
void k_combine(const float* __restrict__ x, const float* __restrict__ agg,
               const float* __restrict__ eps, __bf16* __restrict__ out, int n4) {
    int t = blockIdx.x * 256 + threadIdx.x;
    if (t >= n4) return;
    float e = 2.0f + eps[0];
    float4 xv = ((const float4*)x)[t];
    float4 av = ((const float4*)agg)[t];
    v4bf o;
    o[0] = (__bf16)(e * xv.x + av.x);
    o[1] = (__bf16)(e * xv.y + av.y);
    o[2] = (__bf16)(e * xv.z + av.z);
    o[3] = (__bf16)(e * xv.w + av.w);
    ((v4bf*)out)[t] = o;
}

// comb256[n] = bf16(concat(a[n], b[n]))  (merge layer A-operand)
__global__ __launch_bounds__(256)
void k_concat_bf16(const float* __restrict__ a, const float* __restrict__ b,
                   __bf16* __restrict__ out, int nrows) {
    int t = blockIdx.x * 256 + threadIdx.x;       // nrows * 64 chunks
    if (t >= nrows * 64) return;
    int n = t >> 6, c = t & 63;
    const float* sp = (c < 32) ? (a + (size_t)n * HID + c * 4)
                               : (b + (size_t)n * HID + (c - 32) * 4);
    float4 v = *(const float4*)sp;
    v4bf o;
    o[0] = (__bf16)v.x; o[1] = (__bf16)v.y; o[2] = (__bf16)v.z; o[3] = (__bf16)v.w;
    *(v4bf*)(out + (size_t)n * 256 + c * 4) = o;
}

__global__ __launch_bounds__(256)
void k_cvt_bf16(const float* __restrict__ in, __bf16* __restrict__ out, int n4) {
    int t = blockIdx.x * 256 + threadIdx.x;
    if (t >= n4) return;
    float4 v = ((const float4*)in)[t];
    v4bf o;
    o[0] = (__bf16)v.x; o[1] = (__bf16)v.y; o[2] = (__bf16)v.z; o[3] = (__bf16)v.w;
    ((v4bf*)out)[t] = o;
}

// ------------------------------- WMMA GEMM ---------------------------------
// C[M,Ntiles*16] = act(A[M,K] @ Bt[N,K]^T + bias). One wave per 16x16 tile.
// A/B fragments follow the CDNA5 16-bit layout: lane<16 holds K {0..7,16..23}
// of row/col (lane&15); lane>=16 holds K {8..15,24..31}.
__global__ __launch_bounds__(256)
void k_gemm_wmma(const __bf16* __restrict__ A, const __bf16* __restrict__ Bt,
                 const float* __restrict__ bias,
                 float* __restrict__ outF, __bf16* __restrict__ outB,
                 int Mtiles, int Ntiles, int K, int ldc, int act) {
    int wid = blockIdx.x * (blockDim.x >> 5) + (threadIdx.x >> 5);
    if (wid >= Mtiles * Ntiles) return;           // wave-uniform exit: EXEC stays ~0
    int tm = wid / Ntiles, tn = wid % Ntiles;     // adjacent waves share the A tile
    int lane = threadIdx.x & 31;
    int half = lane >> 4;
    int l15  = lane & 15;

    const __bf16* Arow = A  + (size_t)(tm * 16 + l15) * K + half * 8;
    const __bf16* Brow = Bt + (size_t)(tn * 16 + l15) * K + half * 8;

    v8f acc = {};
    for (int k0 = 0; k0 < K; k0 += 32) {
        union { v16bf v; v8bf h[2]; } ua, ub;
        ua.h[0] = *(const v8bf*)(Arow + k0);
        ua.h[1] = *(const v8bf*)(Arow + k0 + 16);
        ub.h[0] = *(const v8bf*)(Brow + k0);
        ub.h[1] = *(const v8bf*)(Brow + k0 + 16);
        acc = __builtin_amdgcn_wmma_f32_16x16x32_bf16(
                  false, ua.v, false, ub.v, (short)0, acc, false, false);
    }

    int col = tn * 16 + l15;
    float bv = bias ? bias[col] : 0.0f;
    int rbase = tm * 16 + half * 8;               // VGPR r: lanes<16 -> row r, >=16 -> r+8
#pragma unroll
    for (int r = 0; r < 8; ++r) {
        float v = acc[r] + bv;
        if (act == 1)      v = (v > 0.f) ? v : SLOPE * v;   // leaky relu
        else if (act == 2) v = fmaxf(v, 0.f);               // relu
        size_t o = (size_t)(rbase + r) * ldc + col;
        if (outF) outF[o] = v;
        if (outB) outB[o] = (__bf16)v;
    }
}

// ------------------------------- pooling -----------------------------------

__global__ __launch_bounds__(256)
void k_count(const int* __restrict__ batch, float* __restrict__ counts, int N) {
    int t = blockIdx.x * 256 + threadIdx.x;
    if (t < N) atomicAdd(&counts[batch[t]], 1.0f);
}

__global__ __launch_bounds__(256)
void k_pool_feat(const float* __restrict__ x, const int* __restrict__ batch,
                 float* __restrict__ out, int N) {
    int t = blockIdx.x * 256 + threadIdx.x;       // N * 32 chunks
    if (t >= N * 32) return;
    int n = t >> 5, c = t & 31;
    int g = batch[n];
    float4 v = *(const float4*)(x + (size_t)n * HID + c * 4);
    float* o = out + (size_t)g * HID + c * 4;
    atomicAdd(o + 0, v.x);
    atomicAdd(o + 1, v.y);
    atomicAdd(o + 2, v.z);
    atomicAdd(o + 3, v.w);
}

__global__ __launch_bounds__(256)
void k_pool_logits(const float* __restrict__ lg, const int* __restrict__ batch,
                   float* __restrict__ out, int N) {
    int t = blockIdx.x * 256 + threadIdx.x;       // N * NCLS
    if (t >= N * NCLS) return;
    int n = t / NCLS, c = t % NCLS;
    atomicAdd(&out[(size_t)batch[n] * NCLS + c], lg[(size_t)n * 16 + c]);
}

__global__ __launch_bounds__(256)
void k_divide(float* __restrict__ out, const float* __restrict__ counts) {
    int t = blockIdx.x * 256 + threadIdx.x;       // 8192 + 640
    if (t < NGRAPH * HID) {
        out[t] /= fmaxf(counts[t >> 7], 1.0f);
    } else if (t < NGRAPH * HID + NGRAPH * NCLS) {
        int u = t - NGRAPH * HID;
        out[t] /= fmaxf(counts[u / NCLS], 1.0f);
    }
}

// ------------------------------- driver ------------------------------------

extern "C" void kernel_launch(void* const* d_in, const int* in_sizes, int n_in,
                              void* d_out, int out_size, void* d_ws, size_t ws_size,
                              hipStream_t stream) {
    const float* exp_in = (const float*)d_in[0];
    const float* cnv_in = (const float*)d_in[1];
    const int*   eidx   = (const int*)d_in[2];
    const int*   batch  = (const int*)d_in[3];
    const float* eps1e = (const float*)d_in[4];
    const float* W1e = (const float*)d_in[5];  const float* b1e = (const float*)d_in[6];
    const float* eps2e = (const float*)d_in[7];
    const float* W2e = (const float*)d_in[8];  const float* b2e = (const float*)d_in[9];
    const float* eps1c = (const float*)d_in[10];
    const float* W1c = (const float*)d_in[11]; const float* b1c = (const float*)d_in[12];
    const float* eps2c = (const float*)d_in[13];
    const float* W2c = (const float*)d_in[14]; const float* b2c = (const float*)d_in[15];
    const float* Wm  = (const float*)d_in[16]; const float* bm  = (const float*)d_in[17];
    const float* Wp1 = (const float*)d_in[18]; const float* bp1 = (const float*)d_in[19];
    const float* Wp2 = (const float*)d_in[20]; const float* bp2 = (const float*)d_in[21];

    const int E = in_sizes[2] / 2;
    const int* src = eidx;
    const int* dst = eidx + E;

    // ---- workspace carve-up (256B aligned) ----
    char* ws = (char*)d_ws;
    size_t off = 0;
    auto take = [&](size_t bytes) -> char* {
        char* p = ws + off;
        off += (bytes + 255) & ~(size_t)255;
        return p;
    };
    float*  bufA  = (float*)take((size_t)NPNODES * HID * 4);  // exp act; later bf16 X
    float*  bufB  = (float*)take((size_t)NPNODES * HID * 4);  // cnv act; later logits f32
    float*  bufX  = (float*)take((size_t)NPNODES * HID * 4);  // agg accum; later merged X
    __bf16* comb  = (__bf16*)take((size_t)NPNODES * 256 * 2); // bf16 A-operand staging
    __bf16* W1et  = (__bf16*)take(128 * 128 * 2);
    __bf16* W2et  = (__bf16*)take(128 * 128 * 2);
    __bf16* W1ct  = (__bf16*)take(128 * 64 * 2);
    __bf16* W2ct  = (__bf16*)take(128 * 128 * 2);
    __bf16* Wmt   = (__bf16*)take(128 * 256 * 2);
    __bf16* Wp1t  = (__bf16*)take(128 * 128 * 2);
    __bf16* Wp2t  = (__bf16*)take(16 * 128 * 2);
    float*  bp2p  = (float*)take(16 * 4);
    float*  counts = (float*)take(64 * 4);

    float* agg = bufX;    // aggregation accumulator lives where merged X will go

    const int MT = NPNODES / 16;      // 3125 M tiles
    const int N4 = NPNODES * HID / 4; // float4 count of a node matrix

    auto zero = [&](float* p, int n4) {
        k_zero4<<<cdivi(n4, 256), 256, 0, stream>>>(p, n4);
    };
    auto gemm = [&](const __bf16* A, const __bf16* Bt, const float* bias,
                    float* oF, __bf16* oB, int Ntiles, int K, int ldc, int act) {
        int waves = MT * Ntiles;
        k_gemm_wmma<<<cdivi(waves, 8), 256, 0, stream>>>(A, Bt, bias, oF, oB,
                                                         MT, Ntiles, K, ldc, act);
    };

    // ---- output / counts init (harness poisons once; we must zero) ----
    zero((float*)d_out, out_size / 4);
    zero(counts, 64 / 4);

    // ---- weight pre-pack (bf16, transposed for contiguous B fragments) ----
    k_pack_wt<<<cdivi(128 * 128, 256), 256, 0, stream>>>(W1e, W1et, 128, 128, 128);
    k_pack_wt<<<cdivi(128 * 128, 256), 256, 0, stream>>>(W2e, W2et, 128, 128, 128);
    k_pack_wt<<<cdivi(128 * 64,  256), 256, 0, stream>>>(W1c, W1ct,  64, 128, 128);
    k_pack_wt<<<cdivi(128 * 128, 256), 256, 0, stream>>>(W2c, W2ct, 128, 128, 128);
    k_pack_wt<<<cdivi(128 * 256, 256), 256, 0, stream>>>(Wm,  Wmt,  256, 128, 128);
    k_pack_wt<<<cdivi(128 * 128, 256), 256, 0, stream>>>(Wp1, Wp1t, 128, 128, 128);
    k_pack_wt<<<cdivi(16 * 128,  256), 256, 0, stream>>>(Wp2, Wp2t, 128, NCLS, 16);
    k_pad_bias<<<1, 16, 0, stream>>>(bp2, bp2p, NCLS, 16);

    // ---- exp path, GIN layer 1 (K=128) ----
    zero(agg, N4);
    k_scatter<<<cdivi(E * 32, 256), 256, 0, stream>>>(exp_in, src, dst, agg, E, 128);
    k_combine<<<cdivi(N4, 256), 256, 0, stream>>>(exp_in, agg, eps1e, comb, N4);
    gemm(comb, W1et, b1e, bufA, nullptr, 8, 128, HID, /*lrelu*/1);

    // ---- exp path, GIN layer 2 ----
    zero(agg, N4);
    k_scatter<<<cdivi(E * 32, 256), 256, 0, stream>>>(bufA, src, dst, agg, E, 128);
    k_combine<<<cdivi(N4, 256), 256, 0, stream>>>(bufA, agg, eps2e, comb, N4);
    gemm(comb, W2et, b2e, bufA, nullptr, 8, 128, HID, 1);

    // ---- cnv path, GIN layer 1 (K=64) ----
    zero(agg, NPNODES * 64 / 4);
    k_scatter<<<cdivi(E * 16, 256), 256, 0, stream>>>(cnv_in, src, dst, agg, E, 64);
    k_combine<<<cdivi(NPNODES * 16, 256), 256, 0, stream>>>(cnv_in, agg, eps1c, comb,
                                                            NPNODES * 16);
    gemm(comb, W1ct, b1c, bufB, nullptr, 8, 64, HID, 1);

    // ---- cnv path, GIN layer 2 ----
    zero(agg, N4);
    k_scatter<<<cdivi(E * 32, 256), 256, 0, stream>>>(bufB, src, dst, agg, E, 128);
    k_combine<<<cdivi(N4, 256), 256, 0, stream>>>(bufB, agg, eps2c, comb, N4);
    gemm(comb, W2ct, b2c, bufB, nullptr, 8, 128, HID, 1);

    // ---- merge: lrelu(concat(xe, xc) @ Wm + bm) -> X (bufX, agg now dead) ----
    k_concat_bf16<<<cdivi(NPNODES * 64, 256), 256, 0, stream>>>(bufA, bufB, comb, NPNODES);
    gemm(comb, Wmt, bm, bufX, nullptr, 8, 256, HID, 1);

    // ---- graph mean pool of X ----
    k_count<<<cdivi(NPNODES, 256), 256, 0, stream>>>(batch, counts, NPNODES);
    k_pool_feat<<<cdivi(NPNODES * 32, 256), 256, 0, stream>>>(bufX, batch,
                                                              (float*)d_out, NPNODES);

    // ---- MLP head: relu(X@Wp1+bp1) @ Wp2 + bp2 ----
    __bf16* x16 = (__bf16*)bufA;                 // bufA free: X as bf16 A-operand
    k_cvt_bf16<<<cdivi(N4, 256), 256, 0, stream>>>(bufX, x16, N4);
    __bf16* h16 = comb;                          // comb free: hidden written as bf16
    gemm(x16, Wp1t, bp1, nullptr, h16, 8, 128, HID, /*relu*/2);
    float* logits = bufB;                        // bufB free: [NPNODES x 16] f32
    gemm(h16, Wp2t, bp2p, logits, nullptr, 1, 128, 16, /*none*/0);

    // ---- graph mean pool of logits ----
    k_pool_logits<<<cdivi(NPNODES * NCLS, 256), 256, 0, stream>>>(
        logits, batch, (float*)d_out + NGRAPH * HID, NPNODES);

    // ---- divide both pools by counts ----
    k_divide<<<cdivi(NGRAPH * HID + NGRAPH * NCLS, 256), 256, 0, stream>>>(
        (float*)d_out, counts);

    (void)n_in; (void)ws_size; (void)eps1e; (void)eps2e;
}
